// RCNN_65189013619215
// MI455X (gfx1250) — matrix operand order repaired and checked by reading, not compiled, not run on hardware
//
#include <hip/hip_runtime.h>

// ---------------------------------------------------------------------------
// Types / helpers
// ---------------------------------------------------------------------------
typedef __bf16 bf16;
typedef __attribute__((ext_vector_type(16))) __bf16 v16bf;
typedef __attribute__((ext_vector_type(8)))  float  v8f;
typedef int v4i __attribute__((vector_size(16)));
typedef __attribute__((address_space(1))) v4i* gvec_p;   // global v4i*
typedef __attribute__((address_space(3))) v4i* lvec_p;   // LDS v4i*

#define BM 128
#define BN 64
#define BK 32

// CDNA5 async global->LDS staging (ASYNCcnt-tracked), if toolchain exposes it
#if __has_builtin(__builtin_amdgcn_global_load_async_to_lds_b128)
#define ASYNC_A 1
#else
#define ASYNC_A 0
#endif

__device__ __forceinline__ void wait_asynccnt0() {
#if __has_builtin(__builtin_amdgcn_s_wait_asynccnt)
  __builtin_amdgcn_s_wait_asynccnt(0);
#else
  asm volatile("s_wait_asynccnt 0" ::: "memory");
#endif
}

__device__ __forceinline__ bf16 f2bf(float f) {
  union { float f; unsigned u; } in; in.f = f;
  unsigned u = in.u;
  u += 0x7FFFu + ((u >> 16) & 1u);        // round-to-nearest-even
  unsigned short h = (unsigned short)(u >> 16);
  union { unsigned short s; bf16 b; } out; out.s = h;
  return out.b;
}
__device__ __forceinline__ float bf2f(bf16 b) {
  union { bf16 b; unsigned short s; } in; in.b = b;
  union { unsigned u; float f; } out; out.u = ((unsigned)in.s) << 16;
  return out.f;
}
__device__ __forceinline__ v8f v8f_zero() {
  v8f z = {0.f,0.f,0.f,0.f,0.f,0.f,0.f,0.f};
  return z;
}

union Frag { v16bf v; float4 f4[2]; };

// ---------------------------------------------------------------------------
// f32 -> bf16 convert (with optional zero pad tail: indices [n, npad) -> 0)
// ---------------------------------------------------------------------------
__global__ void k_cvt(const float* __restrict__ src, bf16* __restrict__ dst,
                      long long n, long long npad) {
  long long i = (long long)blockIdx.x * 256 + threadIdx.x;
  if (i >= npad) return;
  float v = (i < n) ? src[i] : 0.f;
  dst[i] = f2bf(v);
}

// ---------------------------------------------------------------------------
// Generic WMMA GEMM: C[M,N] = epi(A[M,K] * B[K,N])
//   A,B,C bf16 row-major; K % 32 == 0; N % 16 == 0; M arbitrary.
//   EPI==0: bias + relu + batchnorm    EPI==1: bias + relu
// ---------------------------------------------------------------------------
template<int EPI>
__launch_bounds__(256)
__global__ void k_gemm(const bf16* __restrict__ A, const bf16* __restrict__ B,
                       bf16* __restrict__ C, int M, int N, int K,
                       const float* __restrict__ bias,
                       const float* __restrict__ bng, const float* __restrict__ bnb,
                       const float* __restrict__ bnm, const float* __restrict__ bnv) {
  __shared__ __align__(16) bf16 sA[BM * BK];     // [row][k]      8 KB
  __shared__ __align__(16) bf16 sBT[BN * BK];    // [n][k] (T)    4 KB

  const int tid  = threadIdx.x;
  const int lane = tid & 31;
  const int w    = tid >> 5;       // wave 0..7
  const int wm   = w & 3;          // 4 wave rows  -> rows [wm*32, wm*32+32)
  const int wn   = w >> 2;         // 2 wave cols  -> cols [wn*32, wn*32+32)
  const int m0   = blockIdx.y * BM;
  const int n0   = blockIdx.x * BN;
  const int mrow = lane & 15;
  const int half = lane >> 4;

  v8f acc[2][2];
  acc[0][0] = v8f_zero(); acc[0][1] = v8f_zero();
  acc[1][0] = v8f_zero(); acc[1][1] = v8f_zero();

  for (int kb = 0; kb < K; kb += BK) {
    // ---- stage A tile [BM x BK] : 512 chunks of 8 bf16, 2 per thread ----
    #pragma unroll
    for (int t = 0; t < 2; ++t) {
      int c    = tid + t * 256;
      int row  = c >> 2;
      int col8 = (c & 3) << 3;
#if ASYNC_A
      if (m0 + row < M) {
        __builtin_amdgcn_global_load_async_to_lds_b128(
            (gvec_p)(A + (size_t)(m0 + row) * K + kb + col8),
            (lvec_p)(sA + row * BK + col8),
            0, 0);
      } else {
        *(float4*)(sA + row * BK + col8) = make_float4(0.f, 0.f, 0.f, 0.f);
      }
#else
      float4 val = make_float4(0.f, 0.f, 0.f, 0.f);
      if (m0 + row < M)
        val = *(const float4*)(A + (size_t)(m0 + row) * K + kb + col8);
      *(float4*)(sA + row * BK + col8) = val;
#endif
    }
    // ---- stage B tile transposed [BN x BK] : 256 chunks of 8, 1/thread ----
    {
      int krow = tid >> 3;          // 0..31
      int nb   = (tid & 7) << 3;    // 0..56
      union { float4 f4; bf16 h[8]; } u;
      u.f4 = make_float4(0.f, 0.f, 0.f, 0.f);
      if (n0 + nb < N)              // chunk fully in-range (N%16==0, nb%8==0)
        u.f4 = *(const float4*)(B + (size_t)(kb + krow) * N + n0 + nb);
      #pragma unroll
      for (int j = 0; j < 8; ++j) sBT[(nb + j) * BK + krow] = u.h[j];
    }
    // ---- prefetch next K tile into cache while this one is consumed ----
    if (kb + BK < K) {
      int prow = tid >> 1;                       // 0..127
      if ((tid & 1) == 0 && m0 + prow < M)
        __builtin_prefetch((const void*)(A + (size_t)(m0 + prow) * K + kb + BK), 0, 0);
      if (tid < 32)
        __builtin_prefetch((const void*)(B + (size_t)(kb + BK + tid) * N + n0), 0, 0);
    }
#if ASYNC_A
    wait_asynccnt0();
#endif
    __syncthreads();

    // ---- fragments + 4 WMMAs per wave ----
    Frag af[2], bfr[2];
    #pragma unroll
    for (int s = 0; s < 2; ++s) {
      int rb = wm * 32 + s * 16 + mrow;
      af[s].f4[0] = *(const float4*)(sA + rb * BK + half * 8);        // K e0..7
      af[s].f4[1] = *(const float4*)(sA + rb * BK + 16 + half * 8);   // K e8..15
    }
    #pragma unroll
    for (int t = 0; t < 2; ++t) {
      int nb = wn * 32 + t * 16 + mrow;
      const float4* p = (const float4*)(sBT + nb * BK + half * 8);
      bfr[t].f4[0] = p[0];
      bfr[t].f4[1] = p[2];
    }
    #pragma unroll
    for (int s = 0; s < 2; ++s)
      #pragma unroll
      for (int t = 0; t < 2; ++t)
        acc[s][t] = __builtin_amdgcn_wmma_f32_16x16x32_bf16(
            false, af[s].v, false, bfr[t].v, (short)0, acc[s][t], false, false);
    __syncthreads();
  }

  // ---- epilogue + store ----
  #pragma unroll
  for (int t = 0; t < 2; ++t) {
    int col = n0 + wn * 32 + t * 16 + mrow;
    if (col >= N) continue;
    float bi  = bias[col];
    float scl = 1.f, shf = 0.f;
    if (EPI == 0) {
      float inv = rsqrtf(bnv[col] + 1e-3f);
      scl = bng[col] * inv;
      shf = bnb[col] - bnm[col] * scl;
    }
    #pragma unroll
    for (int s = 0; s < 2; ++s) {
      #pragma unroll
      for (int v = 0; v < 8; ++v) {
        int row = m0 + wm * 32 + s * 16 + (half << 3) + v;
        if (row >= M) continue;
        float z = acc[s][t][v] + bi;
        z = fmaxf(z, 0.f);
        if (EPI == 0) z = scl * z + shf;
        C[(size_t)row * N + col] = f2bf(z);
      }
    }
  }
}

// ---------------------------------------------------------------------------
// conv2 im2col: feat1 [4,64,64,384] bf16 -> col [16384, 3456] (SAME pad)
// ---------------------------------------------------------------------------
__global__ void k_im2col_c2(const bf16* __restrict__ feat1, bf16* __restrict__ col) {
  size_t gid = (size_t)blockIdx.x * 256 + threadIdx.x;
  const size_t total = (size_t)16384 * 432;       // chunks of 8 channels
  if (gid >= total) return;
  int    k8   = (int)(gid % 432);
  size_t row  = gid / 432;
  int kidx = k8 * 8;
  int tap  = kidx / 384, c = kidx % 384;
  int dy = tap / 3, dx = tap % 3;
  int b = (int)(row >> 12), rem = (int)(row & 4095);
  int y = rem >> 6, x = rem & 63;
  int sy = y + dy - 1, sx = x + dx - 1;
  float4 v = make_float4(0.f, 0.f, 0.f, 0.f);
  if (sy >= 0 && sy < 64 && sx >= 0 && sx < 64)
    v = *(const float4*)(feat1 + (((size_t)(b * 64 + sy)) * 64 + sx) * 384 + c);
  *(float4*)(col + gid * 8) = v;
}

// ---------------------------------------------------------------------------
// pad: feat2c [4,64,64,192] -> feat2p [4,69,69,192] (zeros bottom/right)
// ---------------------------------------------------------------------------
__global__ void k_pad(const bf16* __restrict__ src, bf16* __restrict__ dst) {
  size_t gid = (size_t)blockIdx.x * 256 + threadIdx.x;
  const size_t total = (size_t)4 * 69 * 69 * 24;
  if (gid >= total) return;
  int    c8 = (int)(gid % 24);
  size_t p  = gid / 24;
  int x = (int)(p % 69); p /= 69;
  int y = (int)(p % 69);
  int b = (int)(p / 69);
  float4 v = make_float4(0.f, 0.f, 0.f, 0.f);
  if (y < 64 && x < 64)
    v = *(const float4*)(src + (((size_t)(b * 64 + y)) * 64 + x) * 192 + c8 * 8);
  *(float4*)(dst + gid * 8) = v;
}

// ---------------------------------------------------------------------------
// ROI patch gather: feat2p [4,69,69,192] -> acond [n*crop*crop, 192]
// ---------------------------------------------------------------------------
__global__ void k_gather_cond(const bf16* __restrict__ feat, const int* __restrict__ rois,
                              bf16* __restrict__ out, int n, int crop, int mstride) {
  size_t gid = (size_t)blockIdx.x * 256 + threadIdx.x;
  size_t total = (size_t)n * crop * crop * 24;
  if (gid >= total) return;
  int    c8 = (int)(gid % 24);
  size_t p  = gid / 24;
  int px = (int)(p % crop); p /= crop;
  int py = (int)(p % crop); p /= crop;       // p = roi index
  const int* r = rois + p * 3;
  int b = r[0];
  int y = r[1] * mstride + py;
  int x = r[2] * mstride + px;
  float4 v = *(const float4*)(feat + (((size_t)(b * 69 + y)) * 69 + x) * 192 + c8 * 8);
  *(float4*)(out + gid * 8) = v;
}

// ---------------------------------------------------------------------------
// ext im2col: cond [n*crop*crop, 144] -> col [n*9, Kpad] (K = kk*kk*144, zero pad)
// ---------------------------------------------------------------------------
__global__ void k_im2col_ext(const bf16* __restrict__ cond, bf16* __restrict__ col,
                             int n, int crop, int kk, int ks, int Kpad) {
  int kw = Kpad / 8;
  size_t gid = (size_t)blockIdx.x * 256 + threadIdx.x;
  size_t total = (size_t)n * 9 * kw;
  if (gid >= total) return;
  int    k8  = (int)(gid % kw);
  size_t row = gid / kw;
  int kidx = k8 * 8;
  float4 v = make_float4(0.f, 0.f, 0.f, 0.f);
  int Kext = kk * kk * 144;
  if (kidx < Kext) {
    int tap = kidx / 144, c = kidx % 144;
    int ky = tap / kk, kx = tap % kk;
    int    op = (int)(row % 9);
    size_t i  = row / 9;
    int oy = op / 3, ox = op % 3;
    size_t srow = i * crop * crop + (size_t)(oy * ks + ky) * crop + (ox * ks + kx);
    v = *(const float4*)(cond + srow * 144 + c);
  }
  *(float4*)(col + row * (size_t)Kpad + kidx) = v;
}

// ---------------------------------------------------------------------------
// combine: maxpool(cond, kk, ks) 3x3x144  ++  ext 3x3x48 -> feats row [1728]
// ---------------------------------------------------------------------------
__global__ void k_combine(const bf16* __restrict__ cond, const bf16* __restrict__ ext,
                          bf16* __restrict__ feats, int n, int crop, int kk, int ks,
                          int rowoff) {
  size_t gid = (size_t)blockIdx.x * 256 + threadIdx.x;
  size_t total = (size_t)n * 9 * 192;
  if (gid >= total) return;
  int    c  = (int)(gid % 192);
  size_t p  = gid / 192;
  int    op = (int)(p % 9);
  size_t i  = p / 9;
  int py = op / 3, px = op % 3;
  float val;
  if (c < 144) {
    float mx = -3.4e38f;
    for (int ky = 0; ky < kk; ++ky)
      for (int kx = 0; kx < kk; ++kx) {
        size_t srow = i * crop * crop + (size_t)(py * ks + ky) * crop + (px * ks + kx);
        mx = fmaxf(mx, bf2f(cond[srow * 144 + c]));
      }
    val = mx;
  } else {
    val = bf2f(ext[((size_t)i * 9 + op) * 48 + (c - 144)]);
  }
  feats[((size_t)(rowoff + i)) * 1728 + op * 192 + c] = f2bf(val);
}

// ---------------------------------------------------------------------------
// final tiny heads: c2 [N,64] x Wcls3[64,3] + softmax ; r2 [N,256] x Wreg3[256,4]
// ---------------------------------------------------------------------------
__global__ void k_heads(const bf16* __restrict__ c2, const bf16* __restrict__ r2,
                        const float* __restrict__ Wc3, const float* __restrict__ bc3,
                        const float* __restrict__ Wr3, const float* __restrict__ br3,
                        float* __restrict__ out, int Ntot) {
  int r = blockIdx.x * 256 + threadIdx.x;
  if (r >= Ntot) return;
  float l0 = bc3[0], l1 = bc3[1], l2 = bc3[2];
  const bf16* c = c2 + (size_t)r * 64;
  for (int k = 0; k < 64; ++k) {
    float x = bf2f(c[k]);
    l0 += x * Wc3[k * 3 + 0];
    l1 += x * Wc3[k * 3 + 1];
    l2 += x * Wc3[k * 3 + 2];
  }
  float mx = fmaxf(l0, fmaxf(l1, l2));
  float e0 = __expf(l0 - mx), e1 = __expf(l1 - mx), e2 = __expf(l2 - mx);
  float inv = 1.f / (e0 + e1 + e2);
  out[(size_t)r * 3 + 0] = e0 * inv;
  out[(size_t)r * 3 + 1] = e1 * inv;
  out[(size_t)r * 3 + 2] = e2 * inv;

  float g0 = br3[0], g1 = br3[1], g2 = br3[2], g3 = br3[3];
  const bf16* rr = r2 + (size_t)r * 256;
  for (int k = 0; k < 256; ++k) {
    float x = bf2f(rr[k]);
    g0 += x * Wr3[k * 4 + 0];
    g1 += x * Wr3[k * 4 + 1];
    g2 += x * Wr3[k * 4 + 2];
    g3 += x * Wr3[k * 4 + 3];
  }
  float* ro = out + (size_t)Ntot * 3 + (size_t)r * 4;
  ro[0] = g0; ro[1] = g1; ro[2] = g2; ro[3] = g3;
}

// ---------------------------------------------------------------------------
// Host-side helpers
// ---------------------------------------------------------------------------
static inline unsigned cdiv_sz(size_t a, unsigned b) { return (unsigned)((a + b - 1) / b); }

static void cvt(hipStream_t st, const float* src, bf16* dst, size_t n, size_t npad) {
  k_cvt<<<cdiv_sz(npad, 256), 256, 0, st>>>(src, dst, (long long)n, (long long)npad);
}

static void gemm(hipStream_t st, const bf16* A, const bf16* B, bf16* C,
                 int M, int N, int K, const float* bias,
                 const float* g, const float* b, const float* m, const float* v) {
  dim3 grid((N + BN - 1) / BN, (M + BM - 1) / BM);
  if (g) k_gemm<0><<<grid, 256, 0, st>>>(A, B, C, M, N, K, bias, g, b, m, v);
  else   k_gemm<1><<<grid, 256, 0, st>>>(A, B, C, M, N, K, bias, nullptr, nullptr, nullptr, nullptr);
}

// ---------------------------------------------------------------------------
// kernel_launch
// Input index map (insertion-order pytree flatten of setup_inputs()):
//  0:x  1:roi2 2:roi3 3:roi5 4:roi8 5:roi12
//  6:Wc1 7:bc1 8-11:bn1(g,b,m,v) 12:Wc2 13:bc2 14-17:bn2
//  scale blocks at 18/30/42/54/66: Wcond,bcond,bna(4),Wext,bext,bnb(4)
//  78:Wcls1 79:bcls1 80:Wcls2 81:bcls2 82:Wcls3 83:bcls3
//  84:Wreg1 85:breg1 86:Wreg2 87:breg2 88:Wreg3 89:breg3
// ---------------------------------------------------------------------------
extern "C" void kernel_launch(void* const* d_in, const int* in_sizes, int n_in,
                              void* d_out, int out_size, void* d_ws, size_t ws_size,
                              hipStream_t stream) {
  (void)in_sizes; (void)n_in; (void)out_size; (void)ws_size;
  auto F = [&](int i) { return (const float*)d_in[i]; };

  char*  base = (char*)d_ws;
  size_t off  = 0;
  auto   A    = [&](size_t elems) -> bf16* {
    bf16* p = (bf16*)(base + off);
    off = (off + elems * sizeof(bf16) + 255) & ~(size_t)255;
    return p;
  };
  bf16* xbf    = A(12582912);     // [16384,768]
  bf16* wbuf   = A(663552);       // reused weight staging (max = Wc2)
  bf16* feat1  = A(6291456);      // [16384,384]
  bf16* col2   = A(56623104);     // [16384,3456]
  bf16* feat2c = A(3145728);      // [16384,192]
  bf16* feat2p = A(3656448);      // [4,69,69,192]
  bf16* acond  = A(56819712);     // [max 295936, 192]
  bf16* condb  = A(42614784);     // [max 295936, 144]
  bf16* extcol = A(107642880);    // [max 9216, 11680]
  bf16* extout = A(3538944);      // [max 73728, 48]
  bf16* feats  = A(40697856);     // [23552, 1728]
  bf16* c1     = A(6029312);      // [23552, 256]
  bf16* r1     = A(6029312);      // [23552, 256]
  bf16* c2b    = A(1507328);      // [23552, 64]
  bf16* r2b    = A(6029312);      // [23552, 256]

  // ---- conv1: 1x1 768->384 + relu + bn1 ----
  cvt(stream, F(0), xbf, 12582912, 12582912);
  cvt(stream, F(6), wbuf, 768 * 384, 768 * 384);
  gemm(stream, xbf, wbuf, feat1, 16384, 384, 768, F(7), F(8), F(9), F(10), F(11));

  // ---- conv2: 3x3 384->192 SAME + relu + bn2, then pad to 69x69 ----
  k_im2col_c2<<<cdiv_sz((size_t)16384 * 432, 256), 256, 0, stream>>>(feat1, col2);
  cvt(stream, F(12), wbuf, 3456 * 192, 3456 * 192);
  gemm(stream, col2, wbuf, feat2c, 16384, 192, 3456, F(13), F(14), F(15), F(16), F(17));
  k_pad<<<cdiv_sz((size_t)4 * 69 * 69 * 24, 256), 256, 0, stream>>>(feat2c, feat2p);

  // ---- per-scale ROI heads ----
  struct SC { int din, n, crop, ms, kk, ks, rowoff, ib; };
  const SC sc[5] = {
      {1, 8192,  3, 1, 1, 1,     0, 18},
      {2, 8192,  5, 1, 3, 1,  8192, 30},
      {3, 4096,  8, 2, 4, 2, 16384, 42},
      {4, 2048, 12, 3, 6, 3, 20480, 54},
      {5, 1024, 17, 5, 9, 4, 22528, 66},
  };
  for (int s = 0; s < 5; ++s) {
    const SC& q = sc[s];
    const int* rois = (const int*)d_in[q.din];
    const int  crop2 = q.crop * q.crop;
    const size_t Mc = (size_t)q.n * crop2;

    cvt(stream, F(q.ib + 0), wbuf, 192 * 144, 192 * 144);
    k_gather_cond<<<cdiv_sz(Mc * 24, 256), 256, 0, stream>>>(feat2p, rois, acond,
                                                             q.n, q.crop, q.ms);
    gemm(stream, acond, wbuf, condb, (int)Mc, 144, 192, F(q.ib + 1),
         F(q.ib + 2), F(q.ib + 3), F(q.ib + 4), F(q.ib + 5));

    const int Kext = q.kk * q.kk * 144;
    const int Kpad = (Kext + 31) & ~31;
    cvt(stream, F(q.ib + 6), wbuf, (size_t)Kext * 48, (size_t)Kpad * 48);
    k_im2col_ext<<<cdiv_sz((size_t)q.n * 9 * (Kpad / 8), 256), 256, 0, stream>>>(
        condb, extcol, q.n, q.crop, q.kk, q.ks, Kpad);
    gemm(stream, extcol, wbuf, extout, q.n * 9, 48, Kpad, F(q.ib + 7),
         F(q.ib + 8), F(q.ib + 9), F(q.ib + 10), F(q.ib + 11));

    k_combine<<<cdiv_sz((size_t)q.n * 9 * 192, 256), 256, 0, stream>>>(
        condb, extout, feats, q.n, q.crop, q.kk, q.ks, q.rowoff);
  }

  // ---- FC heads ----
  const int NT = 23552;
  cvt(stream, F(78), wbuf, 1728 * 256, 1728 * 256);
  gemm(stream, feats, wbuf, c1, NT, 256, 1728, F(79), nullptr, nullptr, nullptr, nullptr);
  cvt(stream, F(84), wbuf, 1728 * 256, 1728 * 256);
  gemm(stream, feats, wbuf, r1, NT, 256, 1728, F(85), nullptr, nullptr, nullptr, nullptr);
  cvt(stream, F(80), wbuf, 256 * 64, 256 * 64);
  gemm(stream, c1, wbuf, c2b, NT, 64, 256, F(81), nullptr, nullptr, nullptr, nullptr);
  cvt(stream, F(86), wbuf, 256 * 256, 256 * 256);
  gemm(stream, r1, wbuf, r2b, NT, 256, 256, F(87), nullptr, nullptr, nullptr, nullptr);

  k_heads<<<cdiv_sz((size_t)NT, 256), 256, 0, stream>>>(
      c2b, r2b, F(82), F(83), F(88), F(89), (float*)d_out, NT);
}
